// MoELayer_14826227106473
// MI455X (gfx1250) — matrix operand order repaired and checked by reading, not compiled, not run on hardware
//
#include <hip/hip_runtime.h>
#include <hip/hip_bf16.h>
#include <math.h>

typedef __attribute__((ext_vector_type(16))) __bf16 v16bf;
typedef __attribute__((ext_vector_type(8)))  float  v8f;

#define D_MODEL 1024
#define D_FF    4096
#define N_EXP   8
#define T_TOK   4096

__device__ __forceinline__ unsigned int pack2bf(float lo, float hi) {
    union { __bf16 h[2]; unsigned int u; } p;
    p.h[0] = (__bf16)lo; p.h[1] = (__bf16)hi;
    return p.u;
}

// branchless tanh-form GELU (H is stored bf16; approx err ~1e-3 << bf16 quantization)
__device__ __forceinline__ float gelu_f(float x) {
    float u = 0.7978845608028654f * (x + 0.044715f * x * x * x);
    u = fminf(fmaxf(u, -15.f), 15.f);
    float t = 1.f - 2.f / (__expf(2.f * u) + 1.f);
    return 0.5f * x * (1.f + t);
}

// ---------------- gating: softmax(x @ gate_w + gate_b) ----------------
__global__ void gate_kernel(const float* __restrict__ x, const float* __restrict__ gw,
                            const float* __restrict__ gb, float* __restrict__ gate) {
    const int lane = threadIdx.x & 31;
    const int wid  = threadIdx.x >> 5;
    const int t    = blockIdx.x * 8 + wid;
    float acc[8];
#pragma unroll
    for (int e = 0; e < 8; ++e) acc[e] = 0.f;
    const float* xr = x + (size_t)t * D_MODEL;
    for (int d = lane; d < D_MODEL; d += 32) {
        float xv = xr[d];
        const float4* g4 = (const float4*)(gw + d * 8);
        float4 glo = g4[0], ghi = g4[1];
        acc[0] += xv * glo.x; acc[1] += xv * glo.y; acc[2] += xv * glo.z; acc[3] += xv * glo.w;
        acc[4] += xv * ghi.x; acc[5] += xv * ghi.y; acc[6] += xv * ghi.z; acc[7] += xv * ghi.w;
    }
#pragma unroll
    for (int e = 0; e < 8; ++e) {
#pragma unroll
        for (int m = 16; m >= 1; m >>= 1) acc[e] += __shfl_xor(acc[e], m, 32);
    }
    if (lane == 0) {
        float mx = -1e30f;
#pragma unroll
        for (int e = 0; e < 8; ++e) { acc[e] += gb[e]; mx = fmaxf(mx, acc[e]); }
        float s = 0.f;
#pragma unroll
        for (int e = 0; e < 8; ++e) { acc[e] = __expf(acc[e] - mx); s += acc[e]; }
        float inv = 1.f / s;
#pragma unroll
        for (int e = 0; e < 8; ++e) gate[t * 8 + e] = acc[e] * inv;
    }
}

// ---------------- x (f32) -> bf16 ----------------
__global__ void cvt_kernel(const float* __restrict__ x, __bf16* __restrict__ xb) {
    size_t i = ((size_t)blockIdx.x * blockDim.x + threadIdx.x) * 8;
    const float4* p = (const float4*)(x + i);
    float4 a = p[0], b = p[1];
    union { unsigned int u[4]; uint4 v; } o;
    o.u[0] = pack2bf(a.x, a.y);
    o.u[1] = pack2bf(a.z, a.w);
    o.u[2] = pack2bf(b.x, b.y);
    o.u[3] = pack2bf(b.z, b.w);
    *(uint4*)(xb + i) = o.v;
}

// ---------------- GEMM: C = A(bf16)[M x K] * B(f32->bf16)[K x N] ----------------
// Block tile 256(M) x 128(N), BK=32, double-buffered LDS, one barrier/step.
// 512 threads = 16 waves: wave_m(0..7) x 32 rows, wave_n(0..1) x 64 cols
//   -> each wave: 2x4 WMMA frags, 64 accumulator VGPRs (no spills).
// MODE 0: H = gelu(C + b1) (bf16)     MODE 1: out (+)= gate * (C + b2)
template<int MODE>
__global__ __launch_bounds__(512)
void ffn_gemm(const __bf16* __restrict__ A, int lda,
              const float* __restrict__ B, int ldb /* == N */,
              int K, const float* __restrict__ bias,
              __bf16* __restrict__ Hout, float* __restrict__ Out,
              const float* __restrict__ gate, int expert, int init)
{
    __shared__ __bf16 As[2][256 * 40];        // 2 x 20 KB (row stride 40 bf16, 16B aligned)
    __shared__ unsigned int Bs[2][16 * 128];  // 2 x 8 KB; Bs[buf][k>>1][n] = {B[k][n],B[k+1][n]}

    const int tid    = threadIdx.x;
    const int lane   = tid & 31;
    const int wid    = tid >> 5;
    const int lo     = lane & 15;
    const int hi     = lane >> 4;
    const int wave_m = wid & 7;               // 8 x 32 rows
    const int wave_n = wid >> 3;              // 2 x 64 cols

    const int m0 = blockIdx.y * 256;
    const int n0 = blockIdx.x * 128;

    v8f acc[2][4];
#pragma unroll
    for (int mi = 0; mi < 2; ++mi)
#pragma unroll
        for (int ni = 0; ni < 4; ++ni) acc[mi][ni] = {};

    // staging: A tile 256 rows x 32 bf16; each thread: half a row (16 bf16 = 2x uint4)
    const int a_row = tid >> 1;               // 0..255
    const int a_col = (tid & 1) * 16;         // 0 / 16 (bf16 units)
    const __bf16* a_src = A + (size_t)(m0 + a_row) * lda + a_col;
    // staging: B; thread handles K pair row b_p, 4 columns (one uint4 of packed pairs)
    const int b_p  = tid >> 5;                // 0..15
    const int b_nc = (tid & 31) * 4;          // 0..124
    const float* b_src0 = B + (size_t)(2 * b_p)     * ldb + n0 + b_nc;
    const float* b_src1 = B + (size_t)(2 * b_p + 1) * ldb + n0 + b_nc;

    uint4  av0, av1;
    float4 bv0, bv1;

    // ---- prologue: fetch tile 0, stage into buffer 0 ----
    {
        const uint4* ag = (const uint4*)a_src;
        av0 = ag[0]; av1 = ag[1];
        bv0 = *(const float4*)b_src0;
        bv1 = *(const float4*)b_src1;
        *(uint4*)&As[0][a_row * 40 + a_col]     = av0;
        *(uint4*)&As[0][a_row * 40 + a_col + 8] = av1;
        *(uint4*)&Bs[0][b_p * 128 + b_nc] = make_uint4(pack2bf(bv0.x, bv1.x), pack2bf(bv0.y, bv1.y),
                                                       pack2bf(bv0.z, bv1.z), pack2bf(bv0.w, bv1.w));
    }

    for (int k0 = 0; k0 < K; k0 += 32) {
        const int cur = (k0 >> 5) & 1;
        __syncthreads();   // buf[cur] ready; buf[cur^1] no longer being read by anyone

        // ---- prefetch tile k0+32 (overlaps the WMMAs below) ----
        const bool has_next = (k0 + 32) < K;
        if (has_next) {
            const uint4* ag = (const uint4*)(a_src + k0 + 32);
            av0 = ag[0]; av1 = ag[1];
            bv0 = *(const float4*)(b_src0 + (size_t)(k0 + 32) * ldb);
            bv1 = *(const float4*)(b_src1 + (size_t)(k0 + 32) * ldb);
        }

        // ---- compute from buf[cur] ----
        union BF { unsigned int u[8]; v16bf v; };
        BF a_frag[2], b_frag[4];
#pragma unroll
        for (int mi = 0; mi < 2; ++mi) {
            int row = wave_m * 32 + mi * 16 + lo;
            uint4 q0 = *(const uint4*)&As[cur][row * 40 + hi * 16];      // K = hi*16..+7
            uint4 q1 = *(const uint4*)&As[cur][row * 40 + hi * 16 + 8];  // K = hi*16+8..+15
            a_frag[mi].u[0] = q0.x; a_frag[mi].u[1] = q0.y; a_frag[mi].u[2] = q0.z; a_frag[mi].u[3] = q0.w;
            a_frag[mi].u[4] = q1.x; a_frag[mi].u[5] = q1.y; a_frag[mi].u[6] = q1.z; a_frag[mi].u[7] = q1.w;
        }
#pragma unroll
        for (int ni = 0; ni < 4; ++ni) {
            int col = wave_n * 64 + ni * 16 + lo;
#pragma unroll
            for (int v = 0; v < 8; ++v)
                b_frag[ni].u[v] = Bs[cur][(hi * 8 + v) * 128 + col];
        }
#pragma unroll
        for (int mi = 0; mi < 2; ++mi)
#pragma unroll
            for (int ni = 0; ni < 4; ++ni)
                acc[mi][ni] = __builtin_amdgcn_wmma_f32_16x16x32_bf16(
                    false, a_frag[mi].v, false, b_frag[ni].v,
                    (short)0, acc[mi][ni], false, false);

        // ---- stage prefetched tile into buf[cur^1] ----
        if (has_next) {
            const int nxt = cur ^ 1;
            *(uint4*)&As[nxt][a_row * 40 + a_col]     = av0;
            *(uint4*)&As[nxt][a_row * 40 + a_col + 8] = av1;
            *(uint4*)&Bs[nxt][b_p * 128 + b_nc] = make_uint4(pack2bf(bv0.x, bv1.x), pack2bf(bv0.y, bv1.y),
                                                             pack2bf(bv0.z, bv1.z), pack2bf(bv0.w, bv1.w));
        }
    }

    // ---- epilogue; C/D layout: M = v + 8*hi, N = lo (per 16x16 tile) ----
    const int N = ldb;
#pragma unroll
    for (int mi = 0; mi < 2; ++mi) {
#pragma unroll
        for (int v = 0; v < 8; ++v) {
            int row = m0 + wave_m * 32 + mi * 16 + v + hi * 8;
            float g = 0.f;
            if (MODE == 1) g = gate[row * N_EXP + expert];
#pragma unroll
            for (int ni = 0; ni < 4; ++ni) {
                int col = n0 + wave_n * 64 + ni * 16 + lo;
                float val = acc[mi][ni][v] + bias[col];
                if (MODE == 0) {
                    Hout[(size_t)row * N + col] = (__bf16)gelu_f(val);
                } else {
                    float o = g * val;
                    float* op = Out + (size_t)row * N + col;
                    if (init) *op = o; else *op += o;
                }
            }
        }
    }
}

extern "C" void kernel_launch(void* const* d_in, const int* in_sizes, int n_in,
                              void* d_out, int out_size, void* d_ws, size_t ws_size,
                              hipStream_t stream) {
    const float* x  = (const float*)d_in[0];   // [4096, 1024]
    const float* gw = (const float*)d_in[1];   // [1024, 8]
    const float* gb = (const float*)d_in[2];   // [8]
    const float* w1 = (const float*)d_in[3];   // [8, 1024, 4096]
    const float* b1 = (const float*)d_in[4];   // [8, 4096]
    const float* w2 = (const float*)d_in[5];   // [8, 4096, 1024]
    const float* b2 = (const float*)d_in[6];   // [8, 1024]
    float* out = (float*)d_out;                // [4096, 1024]

    char* ws = (char*)d_ws;
    float*  gate = (float*)ws;                                           // 128 KB
    __bf16* xb   = (__bf16*)(ws + 131072);                               // 8 MB
    __bf16* h    = (__bf16*)(ws + 131072 + (size_t)T_TOK * D_MODEL * 2); // 32 MB (per-expert, L2-resident)

    gate_kernel<<<T_TOK / 8, 256, 0, stream>>>(x, gw, gb, gate);
    cvt_kernel<<<(T_TOK * D_MODEL) / 2048, 256, 0, stream>>>(x, xb);

    for (int e = 0; e < N_EXP; ++e) {
        ffn_gemm<0><<<dim3(D_FF / 128, T_TOK / 256), 512, 0, stream>>>(
            xb, D_MODEL, w1 + (size_t)e * D_MODEL * D_FF, D_FF, D_MODEL,
            b1 + (size_t)e * D_FF, h, nullptr, nullptr, e, 0);
        ffn_gemm<1><<<dim3(D_MODEL / 128, T_TOK / 256), 512, 0, stream>>>(
            h, D_FF, w2 + (size_t)e * D_FF * D_MODEL, D_MODEL, D_FF,
            b2 + (size_t)e * D_MODEL, nullptr, out, gate, e, e == 0 ? 1 : 0);
    }
}
